// ReprogrammingLayer_2662879724230
// MI455X (gfx1250) — matrix operand order repaired and checked by reading, not compiled, not run on hardware
//
#include <hip/hip_runtime.h>
#include <hip/hip_bf16.h>

typedef __attribute__((ext_vector_type(16))) __bf16 v16bf;
typedef __attribute__((ext_vector_type(8)))  __bf16 bf16x8;
typedef __attribute__((ext_vector_type(8)))  float  v8f;

#define WMMA_BF16(a, b, c) \
  __builtin_amdgcn_wmma_f32_16x16x32_bf16(false, (a), false, (b), (short)0, (c), false, false)

__device__ __forceinline__ v16bf cat8(bf16x8 lo, bf16x8 hi) {
  return __builtin_shufflevector(lo, hi, 0, 1, 2, 3, 4, 5, 6, 7,
                                 8, 9, 10, 11, 12, 13, 14, 15);
}

__device__ __forceinline__ bf16x8 cvt8(float4 a, float4 b) {
  bf16x8 o;
  o[0] = (__bf16)a.x; o[1] = (__bf16)a.y; o[2] = (__bf16)a.z; o[3] = (__bf16)a.w;
  o[4] = (__bf16)b.x; o[5] = (__bf16)b.y; o[6] = (__bf16)b.z; o[7] = (__bf16)b.w;
  return o;
}

// ---------------------------------------------------------------------------
// Transpose + convert: out[n][k] = (bf16) in[k][n].   in: (K,N) f32 row-major.
// 32x32 tiles through LDS; both global sides coalesced.
// ---------------------------------------------------------------------------
__global__ __launch_bounds__(256) void transpose_cvt(const float* __restrict__ in,
                                                     void* __restrict__ outv,
                                                     int K, int N) {
  __shared__ float t[32][33];
  __bf16* out = (__bf16*)outv;
  const int n0 = blockIdx.x * 32, k0 = blockIdx.y * 32;
  const int tx = threadIdx.x, ty = threadIdx.y;
#pragma unroll
  for (int i = 0; i < 4; ++i)
    t[ty + i * 8][tx] = in[(size_t)(k0 + ty + i * 8) * N + n0 + tx];
  __syncthreads();
#pragma unroll
  for (int i = 0; i < 4; ++i)
    out[(size_t)(n0 + ty + i * 8) * K + k0 + tx] = (__bf16)t[tx][ty + i * 8];
}

// ---------------------------------------------------------------------------
// Streaming WMMA GEMM, no LDS: operands are L2-resident on MI455X (192 MB L2).
//   C = A(MxK) * BT^T + bias,  BT is bf16 (N,K) row-major (pre-transposed W).
//   A: f32 (converted in-register) or bf16, row-major; rows >= M read as zero.
//   OUT_MODE 0: bf16 row-major | 1: bf16 transposed C[n*ldt+m] | 2: f32 rm
// block = 256 (8 waves); tile 128x128; wave = 2 m-tiles x 4 n-tiles.
// ---------------------------------------------------------------------------
template <bool A_BF16, int OUT_MODE>
__global__ __launch_bounds__(256) void gemm_wmma(const void* __restrict__ Av,
                                                 const void* __restrict__ BTv,
                                                 const float* __restrict__ bias,
                                                 void* __restrict__ Cv,
                                                 int M, int K, int N, int ldt) {
  const int tid  = threadIdx.x;
  const int m0   = blockIdx.y * 128;
  const int n0   = blockIdx.x * 128;
  const int lane = tid & 31;
  const int w    = tid >> 5;
  const int r    = lane & 15;
  const int hi   = lane >> 4;
  const int kb8  = hi * 8;    // A-frag K-half base
  const int hi16 = hi * 16;   // unused; B-frag also uses kb8
  const int mtb  = (w & 3) * 2;
  const int ntb  = (w >> 2) * 4;

  const __bf16* BT = (const __bf16*)BTv;
  const v16bf vzero = (v16bf)(__bf16)0.0f;

  const float*  aF[2];
  const __bf16* aB[2];
  bool mval[2];
#pragma unroll
  for (int mi = 0; mi < 2; ++mi) {
    const int row = m0 + (mtb + mi) * 16 + r;
    mval[mi] = row < M;
    aF[mi] = (const float*)Av + (size_t)row * K;
    aB[mi] = (const __bf16*)Av + (size_t)row * K;
  }
  const __bf16* bP[4];
#pragma unroll
  for (int ni = 0; ni < 4; ++ni)
    bP[ni] = BT + (size_t)(n0 + (ntb + ni) * 16 + r) * K;

  v8f acc[2][4] = {};

  for (int k0 = 0; k0 < K; k0 += 32) {
    v16bf af[2];
#pragma unroll
    for (int mi = 0; mi < 2; ++mi) {
      if (mval[mi]) {
        if constexpr (A_BF16) {
          const bf16x8* p = (const bf16x8*)(aB[mi] + k0 + kb8);
          af[mi] = cat8(p[0], p[2]);           // K = kb8..+7 and 16+kb8..+7
        } else {
          const float4* p = (const float4*)(aF[mi] + k0 + kb8);
          af[mi] = cat8(cvt8(p[0], p[1]), cvt8(p[4], p[5]));
        }
      } else {
        af[mi] = vzero;
      }
    }
    v16bf bf[4];
#pragma unroll
    for (int ni = 0; ni < 4; ++ni) {
      const bf16x8* p = (const bf16x8*)(bP[ni] + k0 + kb8);
      bf[ni] = cat8(p[0], p[2]);               // K = kb8..+7 and 16+kb8..+7
    }
#pragma unroll
    for (int mi = 0; mi < 2; ++mi)
#pragma unroll
      for (int ni = 0; ni < 4; ++ni)
        acc[mi][ni] = WMMA_BF16(af[mi], bf[ni], acc[mi][ni]);
  }

  // epilogue: bias + store (C layout: vgpr j -> row j + hi*8, col r)
#pragma unroll
  for (int mi = 0; mi < 2; ++mi)
#pragma unroll
    for (int ni = 0; ni < 4; ++ni) {
      const int gcol = n0 + (ntb + ni) * 16 + r;
      const float bb = bias[gcol];
#pragma unroll
      for (int j = 0; j < 8; ++j) {
        const int grow = m0 + (mtb + mi) * 16 + j + hi * 8;
        const float val = acc[mi][ni][j] + bb;
        if constexpr (OUT_MODE == 0) {
          ((__bf16*)Cv)[(size_t)grow * N + gcol] = (grow < M) ? (__bf16)val : (__bf16)0.0f;
        } else if constexpr (OUT_MODE == 1) {
          ((__bf16*)Cv)[(size_t)gcol * ldt + grow] = (grow < M) ? (__bf16)val : (__bf16)0.0f;
        } else {
          ((float*)Cv)[(size_t)grow * N + gcol] = val;
        }
      }
    }
}

// ---------------------------------------------------------------------------
// Attention: one wave per (b, h, 16 L-rows).
//   Q  : bf16 (B*L, 1024) row-major        (A fragments)
//   Kb : bf16 (1024, 1024) row-major       (B fragments, rows >=1000 zero)
//   Vt : bf16 (1024, 1024) = V^T per head  Vt[(h*128+e)*1024 + s]
//   Oa : bf16 (B*L, 1024) attention output
// Full 16 x 1024 f32 score strip in exactly 64 KB LDS; softmax stats via
// half-wave shuffle; P fragments built on the fly (exp -> bf16).
// ---------------------------------------------------------------------------
__global__ __launch_bounds__(32) void attn_wmma(const void* __restrict__ Qv,
                                                const void* __restrict__ Kv,
                                                const void* __restrict__ Vv,
                                                void* __restrict__ Ov) {
  constexpr int SREAL = 1000;
  constexpr float SCALE = 0.08838834764831845f;  // 1/sqrt(128)
  __shared__ float sc[16 * 1024];

  const __bf16* Q  = (const __bf16*)Qv;
  const __bf16* Kb = (const __bf16*)Kv;
  const __bf16* Vt = (const __bf16*)Vv;
  __bf16*       Oa = (__bf16*)Ov;

  const int lane = threadIdx.x;
  const int r    = lane & 15;
  const int hi   = lane >> 4;
  const int lt   = blockIdx.x;
  const int h    = blockIdx.y;
  const int b    = blockIdx.z;

  const __bf16* qrow = Q + ((size_t)(b * 1024 + lt * 16 + r)) * 1024 + h * 128;
  v16bf aq[4];
#pragma unroll
  for (int c = 0; c < 4; ++c) {
    bf16x8 lo = *(const bf16x8*)(qrow + c * 32 + hi * 8);
    bf16x8 h8 = *(const bf16x8*)(qrow + c * 32 + 16 + hi * 8);
    aq[c] = cat8(lo, h8);
  }

  for (int st = 0; st < 64; ++st) {
    v8f acc = {};
#pragma unroll
    for (int c = 0; c < 4; ++c) {
      const __bf16* kp = Kb + (size_t)(st * 16 + r) * 1024 + h * 128 + c * 32 + hi * 8;
      v16bf bk = cat8(*(const bf16x8*)kp, *(const bf16x8*)(kp + 16));
      acc = WMMA_BF16(aq[c], bk, acc);
    }
#pragma unroll
    for (int j = 0; j < 8; ++j)
      sc[(j + hi * 8) * 1024 + st * 16 + r] = acc[j];
  }
  __syncthreads();

  float mx = -__builtin_inff();
  for (int s = hi; s < SREAL; s += 2)
    mx = fmaxf(mx, sc[r * 1024 + s]);
  mx = fmaxf(mx, __shfl_xor(mx, 16, 32));
  const float msc = mx * SCALE;
  float sum = 0.0f;
  for (int s = hi; s < SREAL; s += 2)
    sum += __expf(SCALE * sc[r * 1024 + s] - msc);
  sum += __shfl_xor(sum, 16, 32);
  const float inv_l = 1.0f / sum;

  v8f acc2[8] = {};
  for (int st = 0; st < 32; ++st) {
    const int sb = st * 32;
    v16bf p;
#pragma unroll
    for (int j = 0; j < 8; ++j) {
      const int s = sb + hi * 8 + j;
      p[j] = (__bf16)((s < SREAL) ? __expf(SCALE * sc[r * 1024 + s] - msc) * inv_l : 0.0f);
    }
#pragma unroll
    for (int j = 0; j < 8; ++j) {
      const int s = sb + 16 + hi * 8 + j;
      p[8 + j] = (__bf16)((s < SREAL) ? __expf(SCALE * sc[r * 1024 + s] - msc) * inv_l : 0.0f);
    }
#pragma unroll
    for (int n = 0; n < 8; ++n) {
      const __bf16* vp = Vt + (size_t)(h * 128 + n * 16 + r) * 1024 + sb + hi * 8;
      v16bf bv = cat8(*(const bf16x8*)vp, *(const bf16x8*)(vp + 16));
      acc2[n] = WMMA_BF16(p, bv, acc2[n]);
    }
  }

#pragma unroll
  for (int n = 0; n < 8; ++n)
#pragma unroll
    for (int j = 0; j < 8; ++j) {
      const int row = b * 1024 + lt * 16 + j + hi * 8;
      const int col = h * 128 + n * 16 + r;
      Oa[(size_t)row * 1024 + col] = (__bf16)acc2[n][j];
    }
}

// ---------------------------------------------------------------------------
extern "C" void kernel_launch(void* const* d_in, const int* in_sizes, int n_in,
                              void* d_out, int out_size, void* d_ws, size_t ws_size,
                              hipStream_t stream) {
  const float* target = (const float*)d_in[0];
  const float* source = (const float*)d_in[1];
  const float* value  = (const float*)d_in[2];
  const float* Wq = (const float*)d_in[3];
  const float* bq = (const float*)d_in[4];
  const float* Wk = (const float*)d_in[5];
  const float* bk = (const float*)d_in[6];
  const float* Wv = (const float*)d_in[7];
  const float* bv = (const float*)d_in[8];
  const float* Wo = (const float*)d_in[9];
  const float* bo = (const float*)d_in[10];
  float* out = (float*)d_out;

  char* ws = (char*)d_ws;
  size_t off = 0;
  auto alloc = [&](size_t bytes) { void* p = ws + off; off += (bytes + 255) & ~(size_t)255; return p; };
  void* WqT = alloc((size_t)1024 * 1024 * 2);   // (N=1024, K=1024) bf16
  void* WkT = alloc((size_t)1024 * 4096 * 2);   // (1024, 4096)
  void* WvT = alloc((size_t)1024 * 4096 * 2);
  void* WoT = alloc((size_t)4096 * 1024 * 2);   // (4096, 1024)
  void* Qb  = alloc((size_t)8192 * 1024 * 2);
  void* Kb  = alloc((size_t)1024 * 1024 * 2);
  void* Vt  = alloc((size_t)1024 * 1024 * 2);
  void* Ao  = alloc((size_t)8192 * 1024 * 2);

  dim3 tb(32, 8);
  transpose_cvt<<<dim3(1024 / 32, 1024 / 32), tb, 0, stream>>>(Wq, WqT, 1024, 1024);
  transpose_cvt<<<dim3(1024 / 32, 4096 / 32), tb, 0, stream>>>(Wk, WkT, 4096, 1024);
  transpose_cvt<<<dim3(1024 / 32, 4096 / 32), tb, 0, stream>>>(Wv, WvT, 4096, 1024);
  transpose_cvt<<<dim3(4096 / 32, 1024 / 32), tb, 0, stream>>>(Wo, WoT, 1024, 4096);

  dim3 blk(256);
  gemm_wmma<false, 0><<<dim3(1024 / 128, 8192 / 128), blk, 0, stream>>>(
      target, WqT, bq, Qb, 8192, 1024, 1024, 0);
  gemm_wmma<false, 0><<<dim3(1024 / 128, 1024 / 128), blk, 0, stream>>>(
      source, WkT, bk, Kb, 1000, 4096, 1024, 0);
  gemm_wmma<false, 1><<<dim3(1024 / 128, 1024 / 128), blk, 0, stream>>>(
      value, WvT, bv, Vt, 1000, 4096, 1024, 1024);

  attn_wmma<<<dim3(64, 8, 8), dim3(32), 0, stream>>>(Qb, Kb, Vt, Ao);

  gemm_wmma<true, 2><<<dim3(4096 / 128, 8192 / 128), blk, 0, stream>>>(
      Ao, WoT, bo, out, 8192, 1024, 4096, 0);
}
